// Phase12ShallowInteractionModel_70248485093776
// MI455X (gfx1250) — matrix operand order repaired and checked by reading, not compiled, not run on hardware
//
#include <hip/hip_runtime.h>
#include <hip/hip_fp16.h>
#include <cstdint>
#include <cstddef>

#define BN 8
#define TN 48000
#define CN 64
#define DN 96
#define KN 256
#define ALPHA_C 0.1f
#define BETA_C 0.25f

typedef __attribute__((ext_vector_type(16))) _Float16 v16h;
typedef __attribute__((ext_vector_type(8)))  float    v8f;
// 4-byte-aligned float4 so dword-aligned (not 16B) vector loads are legal
typedef float f4a __attribute__((ext_vector_type(4), aligned(4)));

__device__ __forceinline__ v8f wmma_f16(v16h a, v16h b, v8f c) {
  return __builtin_amdgcn_wmma_f32_16x16x32_f16(false, a, false, b, (short)0, c, false, false);
}

__device__ __forceinline__ float silu_f(float x)    { return x / (1.f + __expf(-x)); }
__device__ __forceinline__ float sigmoid_f(float x) { return 1.f / (1.f + __expf(-x)); }
__device__ __forceinline__ float softplus_f(float x){ return (x > 20.f) ? x : log1pf(__expf(x)); }

// ---------------------------------------------------------------------------
// Pack a [Co][KK] f32 weight (KK = Ci*k flat) into WMMA A-fragment layout:
// wp[((tile*nch + ch)*32 + lane)*16 + e] , A: lane -> M=lane&15 (= co-co0),
// K = ch*32 + (lane>=16 ? 8:0) + (e<8 ? e : e+8).  Out-of-range -> 0.
// Also used for the codebook (Co=256 codes, KK=96).
// ---------------------------------------------------------------------------
__global__ void pack_w_kernel(const float* __restrict__ w, unsigned short* __restrict__ wp,
                              int Co, int KK)
{
  const int tid  = threadIdx.x;            // 0..511
  const int lane = tid >> 4, e = tid & 15;
  const int ch   = blockIdx.x, tile = blockIdx.y;
  const int nch  = (KK + 31) >> 5;
  const int kk   = ch * 32 + ((lane >= 16) ? 8 : 0) + (e < 8 ? e : e + 8);
  const int co   = tile * 16 + (lane & 15);
  float v = 0.f;
  if (co < Co && kk < KK) v = w[(size_t)co * KK + kk];
  wp[(((size_t)tile * nch + ch) * 32 + lane) * 16 + e] = __half_as_ushort(__float2half(v));
}

// B fragment for conv: lane -> N = time t0 (caller passes t0+lane&15); element e ->
// K = kk + (e<8 ? e : e+8) with kk = ch*32 + (lane>=16 ? 8:0) -> (ci,dk).
template<int KW>
__device__ __forceinline__ v16h load_bfrag(const float* __restrict__ xb, int kk, int KK,
                                           int t0, int pad, bool interior)
{
  v16h bf;
  #pragma unroll
  for (int e = 0; e < 16; ++e) {
    const int k = kk + (e < 8 ? e : e + 8);
    float v = 0.f;
    if (k < KK) {
      const int ci = k / KW, dk = k - ci * KW;
      const int t = t0 + dk - pad;
      if (interior || (t >= 0 && t < TN)) v = xb[(size_t)ci * TN + t];
    }
    bf[e] = (_Float16)v;
  }
  return bf;
}

// ---------------------------------------------------------------------------
// Real conv: one wave -> 16 out-channels x 16 time steps.
// A = packed weights (stationary-ish, tiny, cache-resident), B = activations.
// ---------------------------------------------------------------------------
template<int KW, int ACT>
__global__ __launch_bounds__(128)
void conv_real(const float* __restrict__ x, const unsigned short* __restrict__ wp,
               const float* __restrict__ bias, float* __restrict__ y,
               int Ci, int Co, int pad)
{
  const int wave = threadIdx.x >> 5, lane = threadIdx.x & 31;
  const int hi = lane >> 4, lm = lane & 15;
  const int t0 = (blockIdx.x * 4 + wave) * 16;
  if (t0 >= TN) return;                       // wave-uniform
  const int tileY = blockIdx.y, co0 = tileY * 16, b = blockIdx.z;
  const int KK = Ci * KW, nch = (KK + 31) >> 5;
  const bool interior = (t0 >= pad) && (t0 + 15 + KW - 1 - pad < TN);
  const float* __restrict__ xb = x + (size_t)b * Ci * TN;

  v8f acc = {};
  for (int ch = 0; ch < nch; ++ch) {
    const v16h a = *reinterpret_cast<const v16h*>(wp + (((size_t)tileY * nch + ch) * 32 + lane) * 16);
    const v16h bf = load_bfrag<KW>(xb, ch * 32 + 8 * hi, KK, t0 + lm, pad, interior);
    acc = wmma_f16(a, bf, acc);
  }

  const int t = t0 + lm;                      // D: lane -> N = time, VGPR r -> M = co
  #pragma unroll
  for (int r = 0; r < 8; ++r) {
    const int co = co0 + r + 8 * hi;
    if (co < Co) {
      float v = acc[r] + bias[co];
      if      (ACT == 1) v = silu_f(v);
      else if (ACT == 2) v = sigmoid_f(v);
      else if (ACT == 3) v = softplus_f(v) + 1e-6f;
      y[((size_t)b * Co + co) * TN + t] = v;
    }
  }
}

// ---------------------------------------------------------------------------
// Fused complex conv: yr = wr*zr - wi*zi + br ; yi = wi*zr + wr*zi + bi
// 4 WMMA accumulators, no read-modify-write passes.
// ---------------------------------------------------------------------------
template<int KW>
__global__ __launch_bounds__(128)
void conv_cplx(const float* __restrict__ xr, const float* __restrict__ xi,
               const unsigned short* __restrict__ wpr, const unsigned short* __restrict__ wpi,
               const float* __restrict__ br, const float* __restrict__ bi,
               float* __restrict__ yr, float* __restrict__ yi,
               int Ci, int Co, int pad, int writeImag)
{
  const int wave = threadIdx.x >> 5, lane = threadIdx.x & 31;
  const int hi = lane >> 4, lm = lane & 15;
  const int t0 = (blockIdx.x * 4 + wave) * 16;
  if (t0 >= TN) return;
  const int tileY = blockIdx.y, co0 = tileY * 16, b = blockIdx.z;
  const int KK = Ci * KW, nch = (KK + 31) >> 5;
  const bool interior = (t0 >= pad) && (t0 + 15 + KW - 1 - pad < TN);
  const float* __restrict__ xrb = xr + (size_t)b * Ci * TN;
  const float* __restrict__ xib = xi + (size_t)b * Ci * TN;

  v8f aRR = {}, aII = {}, aIR = {}, aRI = {};
  for (int ch = 0; ch < nch; ++ch) {
    const size_t aoff = (((size_t)tileY * nch + ch) * 32 + lane) * 16;
    const v16h awr = *reinterpret_cast<const v16h*>(wpr + aoff);
    const v16h awi = *reinterpret_cast<const v16h*>(wpi + aoff);
    const int kk = ch * 32 + 8 * hi;
    const v16h bzr = load_bfrag<KW>(xrb, kk, KK, t0 + lm, pad, interior);
    const v16h bzi = load_bfrag<KW>(xib, kk, KK, t0 + lm, pad, interior);
    aRR = wmma_f16(awr, bzr, aRR);
    aRI = wmma_f16(awr, bzi, aRI);
    aIR = wmma_f16(awi, bzr, aIR);
    aII = wmma_f16(awi, bzi, aII);
  }

  const int t = t0 + lm;
  #pragma unroll
  for (int r = 0; r < 8; ++r) {
    const int co = co0 + r + 8 * hi;
    if (co < Co) {
      const size_t oi = ((size_t)b * Co + co) * TN + t;
      yr[oi] = aRR[r] - aII[r] + br[co];
      if (writeImag) yi[oi] = aIR[r] + aRI[r] + bi[co];
    }
  }
}

// ---------------------------------------------------------------------------
// VQ nearest code: A = packed codebook tiles (16 codes x 96), B = feat tile
// (96 x 16 times, loaded ONCE per wave, reused across all 16 code tiles).
// dist = ||c||^2 - 2 x.c  (||x||^2 irrelevant for argmin).
// ---------------------------------------------------------------------------
__global__ __launch_bounds__(128)
void vq_argmin_wmma(const float* __restrict__ feat, const unsigned short* __restrict__ cpack,
                    const float* __restrict__ cn, int* __restrict__ idx)
{
  const int wave = threadIdx.x >> 5, lane = threadIdx.x & 31;
  const int hi = lane >> 4, lm = lane & 15;
  const int t0 = (blockIdx.x * 4 + wave) * 16;
  if (t0 >= TN) return;
  const int b = blockIdx.z;

  v16h Bf[3];
  #pragma unroll
  for (int ch = 0; ch < 3; ++ch) {                 // DN = 96 = 3 exact chunks
    #pragma unroll
    for (int e = 0; e < 16; ++e) {
      const int d = ch * 32 + 8 * hi + (e < 8 ? e : e + 8);
      Bf[ch][e] = (_Float16)feat[((size_t)b * DN + d) * TN + t0 + lm];
    }
  }

  float best = 3.4e38f; int bestn = 0;
  for (int nt = 0; nt < KN / 16; ++nt) {
    v8f acc = {};
    #pragma unroll
    for (int ch = 0; ch < 3; ++ch) {
      const v16h a = *reinterpret_cast<const v16h*>(cpack + (((size_t)nt * 3 + ch) * 32 + lane) * 16);
      acc = wmma_f16(a, Bf[ch], acc);
    }
    #pragma unroll
    for (int r = 0; r < 8; ++r) {
      const int n = nt * 16 + r + 8 * hi;          // D: VGPR r -> code row
      const float dist = cn[n] - 2.f * acc[r];
      if (dist < best) { best = dist; bestn = n; }
    }
  }
  // lanes L and L^16 hold the same time step; merge their code halves
  const float ov = __shfl_xor(best, 16, 32);
  const int   oi = __shfl_xor(bestn, 16, 32);
  if (ov < best || (ov == best && oi < bestn)) { best = ov; bestn = oi; }
  if (hi == 0) idx[(size_t)b * TN + t0 + lm] = bestn;
}

__global__ void cnorm_kernel(const float* __restrict__ code, float* __restrict__ cn) {
  int n = blockIdx.x * blockDim.x + threadIdx.x;
  if (n < KN) {
    float s = 0.f;
    for (int d = 0; d < DN; ++d) { float v = code[(size_t)n * DN + d]; s += v * v; }
    cn[n] = s;
  }
}

__global__ void zero_kernel(float* p) { p[0] = 0.f; }

__global__ void vq_apply_kernel(const float* __restrict__ feat, const float* __restrict__ code,
                                const int* __restrict__ idx, float* __restrict__ mdisc,
                                float* __restrict__ lossacc, int* __restrict__ out_idx)
{
  __shared__ float red[256];
  const int tid = threadIdx.x;
  const size_t i = (size_t)blockIdx.x * 256 + tid;   // i = b*T + t
  float ss = 0.f;
  if (i < (size_t)BN * TN) {
    int b = (int)(i / TN), t = (int)(i - (size_t)b * TN);
    int id = idx[i];
    out_idx[i] = id;
    const float* cb = code + (size_t)id * DN;
    for (int d = 0; d < DN; ++d) {
      float q  = cb[d];
      size_t fi = ((size_t)b * DN + d) * TN + t;
      float xv = feat[fi];
      mdisc[fi] = q;
      float df = q - xv;
      ss += df * df;
    }
  }
  red[tid] = ss; __syncthreads();
  for (int s = 128; s > 0; s >>= 1) { if (tid < s) red[tid] += red[tid + s]; __syncthreads(); }
  if (tid == 0) atomicAdd(lossacc, red[0]);
}

__global__ void finalize_kernel(const float* __restrict__ acc, float* __restrict__ out_loss) {
  float inv = 1.0f / ((float)BN * (float)TN * (float)DN);
  out_loss[0] = acc[0] * inv;
  out_loss[1] = BETA_C * acc[0] * inv;
}

__global__ void maglatent_kernel(const float* __restrict__ zr, const float* __restrict__ zi,
                                 float* __restrict__ mag, float* __restrict__ ptrig)
{
  size_t i = (size_t)blockIdx.x * 256 + threadIdx.x;
  const size_t N2 = (size_t)BN * CN * TN;
  if (i >= N2) return;
  int b = (int)(i / ((size_t)CN * TN));
  size_t rem = i - (size_t)b * CN * TN;
  float r = zr[i], q = zi[i];
  float m = sqrtf(r * r + q * q);
  float c_ = 1.f, s_ = 0.f;
  if (m > 0.f) { float inv = 1.f / m; c_ = r * inv; s_ = q * inv; }
  mag[i] = m;
  size_t base = (size_t)b * 2 * CN * TN;
  ptrig[base + rem]                   = c_;
  ptrig[base + (size_t)CN * TN + rem] = s_;
}

__global__ void pint_kernel(const float* __restrict__ pf, const float* __restrict__ gate,
                            float* __restrict__ pint)
{
  size_t i = (size_t)blockIdx.x * 256 + threadIdx.x;
  if (i < (size_t)BN * DN * TN) pint[i] = pf[i] * (1.f + ALPHA_C * gate[i]);
}

__global__ void polar_kernel(const float* __restrict__ mag_hat, const float* __restrict__ ph2c,
                             float* __restrict__ zhr, float* __restrict__ zhi)
{
  size_t i = (size_t)blockIdx.x * 256 + threadIdx.x;
  const size_t N2 = (size_t)BN * CN * TN;
  if (i >= N2) return;
  int b = (int)(i / ((size_t)CN * TN));
  size_t rem = i - (size_t)b * CN * TN;
  size_t base = (size_t)b * 2 * CN * TN;
  float c_ = ph2c[base + rem];
  float s_ = ph2c[base + (size_t)CN * TN + rem];
  float nrm = sqrtf(c_ * c_ + s_ * s_ + 1e-8f);
  float m = mag_hat[i] / nrm;
  zhr[i] = m * c_;
  zhi[i] = m * s_;
}

// ---------------------------------------------------------------------------
extern "C" void kernel_launch(void* const* d_in, const int* in_sizes, int n_in,
                              void* d_out, int out_size, void* d_ws, size_t ws_size,
                              hipStream_t stream)
{
  (void)in_sizes; (void)n_in; (void)out_size; (void)ws_size;
  const float* in[46];
  for (int i = 0; i < 46; ++i) in[i] = (const float*)d_in[i];
  const float* wav = in[0];

  // ---- fp32 activation workspace ----
  const size_t S1 = (size_t)BN * 32  * TN;
  const size_t S2 = (size_t)BN * 64  * TN;
  const size_t S3 = (size_t)BN * 96  * TN;
  const size_t S4 = (size_t)BN * 128 * TN;
  float* W = (float*)d_ws; size_t o = 0;
  float* zr1  = W + o; o += S1;   float* zi1  = W + o; o += S1;
  float* zr2  = W + o; o += S2;   float* zi2  = W + o; o += S2;
  float* zr3  = W + o; o += S2;   float* zi3  = W + o; o += S2;
  float* mag  = W + o; o += S2;
  float* ptrig= W + o; o += S4;
  float* feat1= W + o; o += S3;   float* feat2= W + o; o += S3;
  float* mdisc= W + o; o += S3;
  float* pf1  = W + o; o += S3;   float* pf2  = W + o; o += S3;
  float* cn   = W + o; o += 256;
  float* lossacc = W + o; o += 8;
  int*   idxbuf  = (int*)(W + o); o += (size_t)BN * TN;
  // liveness aliases
  float* tA = feat1;  float* gate = feat2;  float* pintb = pf1;  float* tB = pf2;
  float* mag_hat = mag; float* ph2c = ptrig;
  float* zhr0 = zr2; float* zhi0 = zi2;
  float* dr1 = zr3;  float* di1 = zi3;
  float* dr2 = zr1;  float* di2 = zi1;

  // ---- packed-weight region (f16 fragments), 32B aligned ----
  o = (o + 7) & ~(size_t)7;
  unsigned short* HP = (unsigned short*)(W + o); size_t ho = 0;
  auto pack = [&](const float* w, int Co, int KK) {
    unsigned short* p = HP + ho;
    ho += (size_t)((Co + 15) / 16) * ((KK + 31) / 32) * 512;
    dim3 g((KK + 31) / 32, (Co + 15) / 16);
    pack_w_kernel<<<g, 512, 0, stream>>>(w, p, Co, KK);
    return p;
  };
  unsigned short* p_e1r = pack(in[1], 32, 7);   unsigned short* p_e1i = pack(in[2], 32, 7);
  unsigned short* p_e2r = pack(in[5], 64, 160); unsigned short* p_e2i = pack(in[6], 64, 160);
  unsigned short* p_e3r = pack(in[9], 64, 192); unsigned short* p_e3i = pack(in[10], 64, 192);
  unsigned short* p_d1r = pack(in[13], 64, 192); unsigned short* p_d1i = pack(in[14], 64, 192);
  unsigned short* p_d2r = pack(in[17], 32, 320); unsigned short* p_d2i = pack(in[18], 32, 320);
  unsigned short* p_d3r = pack(in[21], 1, 224);  unsigned short* p_d3i = pack(in[22], 1, 224);
  unsigned short* p_mp1 = pack(in[25], 96, 192);
  unsigned short* p_mp2 = pack(in[27], 96, 288);
  unsigned short* p_ph1 = pack(in[29], 96, 384);
  unsigned short* p_ph2 = pack(in[31], 96, 288);
  unsigned short* p_ia1 = pack(in[33], 96, 96);
  unsigned short* p_ia2 = pack(in[35], 96, 96);
  unsigned short* p_fm1 = pack(in[37], 96, 288);
  unsigned short* p_fm2 = pack(in[39], 64, 96);
  unsigned short* p_fp1 = pack(in[41], 96, 288);
  unsigned short* p_fp2 = pack(in[43], 128, 96);
  unsigned short* p_cb  = pack(in[45], KN, DN);

  float* out_wav  = (float*)d_out;
  int*   out_idx  = (int*)((float*)d_out + (size_t)BN * TN);
  float* out_loss = (float*)d_out + 2 * (size_t)BN * TN;

  const dim3 blk(128);
  auto cgrid = [](int Co) { return dim3(TN / 64, (Co + 15) / 16, BN); };
  const int GB2 = (int)(((size_t)BN * CN * TN + 255) / 256);
  const int GB3 = (int)(((size_t)BN * DN * TN + 255) / 256);

  // ---- encoders ----
  conv_real<7,0><<<cgrid(32), blk, 0, stream>>>(wav, p_e1r, in[3], zr1, 1, 32, 3);
  conv_real<7,0><<<cgrid(32), blk, 0, stream>>>(wav, p_e1i, in[4], zi1, 1, 32, 3);
  conv_cplx<5><<<cgrid(64), blk, 0, stream>>>(zr1, zi1, p_e2r, p_e2i, in[7], in[8], zr2, zi2, 32, 64, 2, 1);
  conv_cplx<3><<<cgrid(64), blk, 0, stream>>>(zr2, zi2, p_e3r, p_e3i, in[11], in[12], zr3, zi3, 64, 64, 1, 1);

  maglatent_kernel<<<GB2, 256, 0, stream>>>(zr3, zi3, mag, ptrig);

  // ---- magnitude branch + VQ ----
  conv_real<3,1><<<cgrid(96), blk, 0, stream>>>(mag,   p_mp1, in[26], feat1, 64, 96, 1);
  conv_real<3,1><<<cgrid(96), blk, 0, stream>>>(feat1, p_mp2, in[28], feat2, 96, 96, 1);
  cnorm_kernel<<<1, 256, 0, stream>>>(in[45], cn);
  vq_argmin_wmma<<<dim3(TN / 64, 1, BN), blk, 0, stream>>>(feat2, p_cb, cn, idxbuf);
  zero_kernel<<<1, 1, 0, stream>>>(lossacc);
  vq_apply_kernel<<<(int)(((size_t)BN * TN + 255) / 256), 256, 0, stream>>>(
      feat2, in[45], idxbuf, mdisc, lossacc, out_idx);

  // ---- phase branch ----
  conv_real<3,1><<<cgrid(96), blk, 0, stream>>>(ptrig, p_ph1, in[30], pf1, 128, 96, 1);
  conv_real<3,1><<<cgrid(96), blk, 0, stream>>>(pf1,   p_ph2, in[32], pf2,  96, 96, 1);

  // ---- gate ----
  conv_real<1,1><<<cgrid(96), blk, 0, stream>>>(mdisc, p_ia1, in[34], tA,   96, 96, 0);
  conv_real<1,2><<<cgrid(96), blk, 0, stream>>>(tA,    p_ia2, in[36], gate, 96, 96, 0);
  pint_kernel<<<GB3, 256, 0, stream>>>(pf2, gate, pintb);

  // ---- heads ----
  conv_real<3,1><<<cgrid(96), blk, 0, stream>>>(mdisc, p_fm1, in[38], tA,      96, 96, 1);
  conv_real<1,3><<<cgrid(64), blk, 0, stream>>>(tA,    p_fm2, in[40], mag_hat, 96, 64, 0);
  conv_real<3,1><<<cgrid(96), blk, 0, stream>>>(pintb, p_fp1, in[42], tB,      96, 96, 1);
  conv_real<1,0><<<cgrid(128), blk, 0, stream>>>(tB,   p_fp2, in[44], ph2c,    96, 128, 0);

  polar_kernel<<<GB2, 256, 0, stream>>>(mag_hat, ph2c, zhr0, zhi0);

  // ---- decoders ----
  conv_cplx<3><<<cgrid(64), blk, 0, stream>>>(zhr0, zhi0, p_d1r, p_d1i, in[15], in[16], dr1, di1, 64, 64, 1, 1);
  conv_cplx<5><<<cgrid(32), blk, 0, stream>>>(dr1,  di1,  p_d2r, p_d2i, in[19], in[20], dr2, di2, 64, 32, 2, 1);
  conv_cplx<7><<<cgrid(1),  blk, 0, stream>>>(dr2,  di2,  p_d3r, p_d3i, in[23], in[24], out_wav, nullptr, 32, 1, 3, 0);

  finalize_kernel<<<1, 1, 0, stream>>>(lossacc, out_loss);
}
